// SwinTransformerBlock_82300163326495
// MI455X (gfx1250) — matrix operand order repaired and checked by reading, not compiled, not run on hardware
//
#include <hip/hip_runtime.h>
#include <hip/hip_bf16.h>
#include <math.h>

// ---------------- types ----------------
typedef __bf16 bf16_t;
typedef __attribute__((ext_vector_type(16))) __bf16 bf16x16;
typedef __attribute__((ext_vector_type(8)))  float   f32x8;
typedef __attribute__((ext_vector_type(4)))  int     i32x4;

union FragAB { bf16x16 v; uint4 q[2]; };

__device__ inline bf16_t f2bf(float x) { return (bf16_t)x; }

__device__ inline f32x8 zero_f32x8() {
  f32x8 z;
#pragma unroll
  for (int i = 0; i < 8; ++i) z[i] = 0.0f;
  return z;
}

// ---------------- CDNA5 async global->LDS copy (ASYNCcnt-tracked) ----------------
#if defined(__HIP_DEVICE_COMPILE__) && defined(__gfx1250__) && \
    __has_builtin(__builtin_amdgcn_global_load_async_to_lds_b128)
#define HAS_ASYNC_LDS 1
#else
#define HAS_ASYNC_LDS 0
#endif

typedef __attribute__((address_space(1))) i32x4 gas_i32x4;  // global int4
typedef __attribute__((address_space(3))) i32x4 las_i32x4;  // LDS int4

__device__ inline void copy_b128(const void* g, void* l) {
#if HAS_ASYNC_LDS
  __builtin_amdgcn_global_load_async_to_lds_b128(
      (gas_i32x4*)(uintptr_t)g, (las_i32x4*)l, 0, 0);
#else
  *(uint4*)l = *(const uint4*)g;
#endif
}

__device__ inline void wait_async_copies() {
#if HAS_ASYNC_LDS
#if __has_builtin(__builtin_amdgcn_s_wait_asynccnt)
  __builtin_amdgcn_s_wait_asynccnt(0);
#else
  asm volatile("s_wait_asynccnt 0" ::: "memory");
#endif
#endif
}

// ---------------- problem constants ----------------
#define C_DIM   768
#define NHEADS  24
#define DH      32
#define WSZ     8
#define SHIFT_  4
#define HID_DIM 3072
#define IMG     96
#define NPIX    (IMG * IMG)       // 9216 tokens
#define NWIN    (12 * 12)         // 144 windows
#define WTOK    (WSZ * WSZ)       // 64 tokens / window
#define EPS_LN  1e-5f

// ---------------- fp32 -> bf16 convert ----------------
__global__ __launch_bounds__(256) void cvt_f32_bf16(const float* __restrict__ src,
                                                    bf16_t* __restrict__ dst, int n) {
  int i = blockIdx.x * 256 + threadIdx.x;
  if (i < n) dst[i] = f2bf(src[i]);
}

// ---------------- LayerNorm over channels (C,H,W input) ----------------
// 32 pixels per block, 8 channel-groups. If `shifted` != 0, the normalized
// value for source pixel (gh,gw) is written to the window-partitioned row of
// the (-SHIFT,-SHIFT)-rolled image; otherwise to plain token order.
__global__ __launch_bounds__(256) void ln_kernel(const float* __restrict__ x,
                                                 const float* __restrict__ g,
                                                 const float* __restrict__ b,
                                                 bf16_t* __restrict__ out,
                                                 int shifted) {
  __shared__ float rs[8][32], rs2[8][32];
  const int px = threadIdx.x & 31;
  const int cg = threadIdx.x >> 5;
  const int p  = blockIdx.x * 32 + px;

  float s = 0.f, s2 = 0.f;
  for (int c = cg; c < C_DIM; c += 8) {
    float v = x[(size_t)c * NPIX + p];
    s += v; s2 += v * v;
  }
  rs[cg][px] = s; rs2[cg][px] = s2;
  __syncthreads();
  float tot = 0.f, tot2 = 0.f;
#pragma unroll
  for (int i = 0; i < 8; ++i) { tot += rs[i][px]; tot2 += rs2[i][px]; }
  const float mean = tot * (1.0f / C_DIM);
  const float var  = tot2 * (1.0f / C_DIM) - mean * mean;
  const float rstd = rsqrtf(var + EPS_LN);

  // destination row
  int row;
  if (shifted) {
    const int gh = p / IMG, gw = p % IMG;
    const int h = (gh + IMG - SHIFT_) % IMG;   // inverse of roll(-SHIFT)
    const int w = (gw + IMG - SHIFT_) % IMG;
    row = ((h >> 3) * 12 + (w >> 3)) * WTOK + (h & 7) * WSZ + (w & 7);
  } else {
    row = p;
  }
  bf16_t* o = out + (size_t)row * C_DIM;
  for (int c = cg; c < C_DIM; c += 8) {
    float v = x[(size_t)c * NPIX + p];
    o[c] = f2bf((v - mean) * rstd * g[c] + b[c]);
  }
}

// ---------------- tiled WMMA GEMM:  out = A(MxK) @ W(NoutxK)^T + bias ----------------
// Double-buffered LDS; global->LDS traffic issued asynchronously (ASYNCcnt) one
// K-step ahead of the WMMA consumption; one barrier per K-step.
// MODE 0: scatter to q/k/v (v transposed) bf16
// MODE 1: proj: window-merge + reverse shift + residual -> f32 (C,H,W)
// MODE 2: fc1:  tanh-GELU -> bf16 row-major (M x Nout)
// MODE 3: fc2:  + residual -> f32 (C,H,W)
template <int MODE>
__global__ __launch_bounds__(256) void wmma_gemm(
    const bf16_t* __restrict__ A, const bf16_t* __restrict__ Bw,
    const float* __restrict__ bias, int K, int Nout,
    float* __restrict__ out_f32, const float* __restrict__ residual,
    bf16_t* __restrict__ out_bf,
    bf16_t* __restrict__ qo, bf16_t* __restrict__ ko, bf16_t* __restrict__ vo) {
  __shared__ bf16_t sA[2][128][40];   // +8 pad: conflict-free b128 fragment reads
  __shared__ bf16_t sB[2][128][40];

  const int tid  = threadIdx.x;
  const int lane = tid & 31;
  const int wv   = tid >> 5;
  const int wm   = wv & 3;          // wave 32-row band
  const int wn   = wv >> 2;         // wave 64-col band
  const int ln   = lane & 15;
  const int kbA  = (lane < 16) ? 0 : 8;
  const int kbB  = (lane < 16) ? 0 : 8;
  const int bm   = blockIdx.x * 128;
  const int bn   = blockIdx.y * 128;

  f32x8 acc[2][4];
#pragma unroll
  for (int i = 0; i < 2; ++i)
#pragma unroll
    for (int j = 0; j < 4; ++j) acc[i][j] = zero_f32x8();

  const int arow  = tid >> 1;
  const int ahalf = (tid & 1) * 16;
  const bf16_t* agBase = A  + (size_t)(bm + arow) * K + ahalf;
  const bf16_t* bgBase = Bw + (size_t)(bn + arow) * K + ahalf;

  // issue one K-step tile (A and B) into LDS buffer `buf`
  auto issue_tile = [&](int buf, int kt) {
    const bf16_t* ag = agBase + kt;
    const bf16_t* bg = bgBase + kt;
    copy_b128(ag,     &sA[buf][arow][ahalf]);
    copy_b128(ag + 8, &sA[buf][arow][ahalf + 8]);
    copy_b128(bg,     &sB[buf][arow][ahalf]);
    copy_b128(bg + 8, &sB[buf][arow][ahalf + 8]);
  };

  const int nsteps = K >> 5;
  issue_tile(0, 0);

  for (int it = 0; it < nsteps; ++it) {
    wait_async_copies();     // own async copies done
    __syncthreads();         // everyone's copies visible; prev buffer free
    if (it + 1 < nsteps) issue_tile((it + 1) & 1, (it + 1) << 5);

    const int buf = it & 1;
    FragAB fa[2], fb[4];
#pragma unroll
    for (int mt = 0; mt < 2; ++mt) {
      const int r = wm * 32 + mt * 16 + ln;
      fa[mt].q[0] = *(const uint4*)&sA[buf][r][kbA];
      fa[mt].q[1] = *(const uint4*)&sA[buf][r][16 + kbA];
    }
#pragma unroll
    for (int nt = 0; nt < 4; ++nt) {
      const int r = wn * 64 + nt * 16 + ln;
      fb[nt].q[0] = *(const uint4*)&sB[buf][r][kbB];
      fb[nt].q[1] = *(const uint4*)&sB[buf][r][kbB + 16];
    }
#pragma unroll
    for (int mt = 0; mt < 2; ++mt)
#pragma unroll
      for (int nt = 0; nt < 4; ++nt)
        acc[mt][nt] = __builtin_amdgcn_wmma_f32_16x16x32_bf16(
            false, fa[mt].v, false, fb[nt].v, (short)0, acc[mt][nt], false, false);
  }

  // ---- epilogue ----
  const int mhalf = (lane >= 16) ? 8 : 0;
#pragma unroll
  for (int mt = 0; mt < 2; ++mt) {
#pragma unroll
    for (int nt = 0; nt < 4; ++nt) {
      const int n = bn + wn * 64 + nt * 16 + ln;
      const float bia = bias[n];
#pragma unroll
      for (int rr = 0; rr < 8; ++rr) {
        const int m = bm + wm * 32 + mt * 16 + rr + mhalf;
        float val = acc[mt][nt][rr] + bia;
        if (MODE == 0) {
          const int part = n / C_DIM;
          const int cc   = n - part * C_DIM;
          const int hh   = cc >> 5, d = cc & 31;
          const int w_   = m >> 6, tk = m & 63;
          if (part == 0)
            qo[(((size_t)w_ * NHEADS + hh) * WTOK + tk) * DH + d] = f2bf(val);
          else if (part == 1)
            ko[(((size_t)w_ * NHEADS + hh) * WTOK + tk) * DH + d] = f2bf(val);
          else
            vo[(((size_t)w_ * NHEADS + hh) * DH + d) * WTOK + tk] = f2bf(val);
        } else if (MODE == 1) {
          const int w_ = m >> 6, tk = m & 63;
          const int h  = (w_ / 12) * WSZ + (tk >> 3);
          const int w2 = (w_ % 12) * WSZ + (tk & 7);
          const int gh = (h + SHIFT_) % IMG;     // reverse shift (roll +SHIFT)
          const int gw = (w2 + SHIFT_) % IMG;
          const size_t idx = (size_t)n * NPIX + gh * IMG + gw;
          out_f32[idx] = residual[idx] + val;
        } else if (MODE == 2) {
          const float u = val;
          const float gl = 0.5f * u * (1.0f + tanhf(0.7978845608028654f *
                                                    (u + 0.044715f * u * u * u)));
          out_bf[(size_t)m * Nout + n] = f2bf(gl);
        } else {  // MODE 3
          const size_t idx = (size_t)n * NPIX + m;
          out_f32[idx] = residual[idx] + val;
        }
      }
    }
  }
}

// ---------------- windowed attention: one block per (window, head) ----------------
__device__ inline int reg96(int t) { return (t < IMG - WSZ) ? 0 : ((t < IMG - SHIFT_) ? 1 : 2); }

__global__ __launch_bounds__(128) void attn_kernel(
    const bf16_t* __restrict__ q, const bf16_t* __restrict__ k,
    const bf16_t* __restrict__ v, const float* __restrict__ bias_table,
    bf16_t* __restrict__ attn_out) {
  __shared__ bf16_t sQ[64][40];
  __shared__ bf16_t sK[64][40];
  __shared__ bf16_t sV[32][72];   // v^T : [dh][token]
  __shared__ bf16_t sP[64][72];   // softmax probs
  __shared__ float  sBias[225];

  const int blk = blockIdx.x;
  const int w_  = blk / NHEADS;
  const int hh  = blk - w_ * NHEADS;
  const int tid  = threadIdx.x;
  const int lane = tid & 31;
  const int wv   = tid >> 5;            // 4 waves: one 16-row band each
  const int ln   = lane & 15;
  const int kbA  = (lane < 16) ? 0 : 8;
  const int kbB  = (lane < 16) ? 0 : 8;
  const int mhalf = (lane >= 16) ? 8 : 0;

  // ---- stage q, k, v^T, bias column ----
  {
    const bf16_t* qg = q + (size_t)blk * (WTOK * DH);
    const bf16_t* kg = k + (size_t)blk * (WTOK * DH);
    const bf16_t* vg = v + (size_t)blk * (DH * WTOK);
    const int r2 = tid >> 1, h2 = (tid & 1) * 16;
    *(uint4*)&sQ[r2][h2]     = ((const uint4*)(qg + r2 * DH + h2))[0];
    *(uint4*)&sQ[r2][h2 + 8] = ((const uint4*)(qg + r2 * DH + h2))[1];
    *(uint4*)&sK[r2][h2]     = ((const uint4*)(kg + r2 * DH + h2))[0];
    *(uint4*)&sK[r2][h2 + 8] = ((const uint4*)(kg + r2 * DH + h2))[1];
    const int r4 = tid >> 2, q4 = (tid & 3) * 16;
    *(uint4*)&sV[r4][q4]     = ((const uint4*)(vg + r4 * WTOK + q4))[0];
    *(uint4*)&sV[r4][q4 + 8] = ((const uint4*)(vg + r4 * WTOK + q4))[1];
    for (int i = tid; i < 225; i += 128) sBias[i] = bias_table[i * NHEADS + hh];
  }
  __syncthreads();

  // ---- S = (q * scale) @ k^T + bias + mask ----
  FragAB fq;
  {
    const int r = wv * 16 + ln;
    fq.q[0] = *(const uint4*)&sQ[r][kbA];
    fq.q[1] = *(const uint4*)&sQ[r][16 + kbA];
  }
  f32x8 sc[4];
#pragma unroll
  for (int nt = 0; nt < 4; ++nt) {
    FragAB fk;
    const int r = nt * 16 + ln;
    fk.q[0] = *(const uint4*)&sK[r][kbB];
    fk.q[1] = *(const uint4*)&sK[r][kbB + 16];
    sc[nt] = __builtin_amdgcn_wmma_f32_16x16x32_bf16(
        false, fq.v, false, fk.v, (short)0, zero_f32x8(), false, false);
  }

  const float scale = 0.17677669529663687f;  // 1/sqrt(32)
  const int whh = w_ / 12, www = w_ % 12;
  int idn[4], jh[4], jw[4];
#pragma unroll
  for (int nt = 0; nt < 4; ++nt) {
    const int n = nt * 16 + ln;
    jh[nt] = n >> 3; jw[nt] = n & 7;
    idn[nt] = reg96(whh * WSZ + jh[nt]) * 3 + reg96(www * WSZ + jw[nt]);
  }

#pragma unroll
  for (int rr = 0; rr < 8; ++rr) {
    const int m  = wv * 16 + mhalf + rr;
    const int ih = m >> 3, iw = m & 7;
    const int idm = reg96(whh * WSZ + ih) * 3 + reg96(www * WSZ + iw);
    float mx = -1e30f;
#pragma unroll
    for (int nt = 0; nt < 4; ++nt) {
      const int bidx = (ih - jh[nt] + 7) * 15 + (iw - jw[nt] + 7);
      float s = sc[nt][rr] * scale + sBias[bidx] + ((idm != idn[nt]) ? -100.0f : 0.0f);
      sc[nt][rr] = s;
      mx = fmaxf(mx, s);
    }
#pragma unroll
    for (int off = 1; off < 16; off <<= 1) mx = fmaxf(mx, __shfl_xor(mx, off, 32));
    float ssum = 0.f;
#pragma unroll
    for (int nt = 0; nt < 4; ++nt) {
      float e = __expf(sc[nt][rr] - mx);
      sc[nt][rr] = e;
      ssum += e;
    }
#pragma unroll
    for (int off = 1; off < 16; off <<= 1) ssum += __shfl_xor(ssum, off, 32);
    const float inv = 1.0f / ssum;
#pragma unroll
    for (int nt = 0; nt < 4; ++nt)
      sP[m][nt * 16 + ln] = f2bf(sc[nt][rr] * inv);
  }
  __syncthreads();

  // ---- O = P @ v  (B operand is v^T rows in sV) ----
  f32x8 oacc[2] = {zero_f32x8(), zero_f32x8()};
#pragma unroll
  for (int ks = 0; ks < 2; ++ks) {
    FragAB fp;
    const int r = wv * 16 + ln;
    fp.q[0] = *(const uint4*)&sP[r][ks * 32 + kbA];
    fp.q[1] = *(const uint4*)&sP[r][ks * 32 + 16 + kbA];
#pragma unroll
    for (int nt = 0; nt < 2; ++nt) {
      FragAB fv;
      const int br = nt * 16 + ln;
      fv.q[0] = *(const uint4*)&sV[br][ks * 32 + kbB];
      fv.q[1] = *(const uint4*)&sV[br][ks * 32 + kbB + 16];
      oacc[nt] = __builtin_amdgcn_wmma_f32_16x16x32_bf16(
          false, fp.v, false, fv.v, (short)0, oacc[nt], false, false);
    }
  }

  // ---- write (window-token row, C) bf16 for the proj GEMM ----
#pragma unroll
  for (int nt = 0; nt < 2; ++nt) {
    const int d = nt * 16 + ln;
#pragma unroll
    for (int rr = 0; rr < 8; ++rr) {
      const int m = wv * 16 + mhalf + rr;
      attn_out[((size_t)w_ * WTOK + m) * C_DIM + hh * DH + d] = f2bf(oacc[nt][rr]);
    }
  }
}

// ---------------- host launcher ----------------
extern "C" void kernel_launch(void* const* d_in, const int* in_sizes, int n_in,
                              void* d_out, int out_size, void* d_ws, size_t ws_size,
                              hipStream_t stream) {
  const float* x_chw  = (const float*)d_in[0];
  const float* n1w    = (const float*)d_in[1];
  const float* n1b    = (const float*)d_in[2];
  const float* qkv_w  = (const float*)d_in[3];
  const float* qkv_b  = (const float*)d_in[4];
  const float* proj_w = (const float*)d_in[5];
  const float* proj_b = (const float*)d_in[6];
  const float* btab   = (const float*)d_in[7];
  const float* n2w    = (const float*)d_in[8];
  const float* n2b    = (const float*)d_in[9];
  const float* fc1_w  = (const float*)d_in[10];
  const float* fc1_b  = (const float*)d_in[11];
  const float* fc2_w  = (const float*)d_in[12];
  const float* fc2_b  = (const float*)d_in[13];
  float* out = (float*)d_out;

  // workspace layout (all sizes are multiples of 256B)
  char* ws = (char*)d_ws;
  bf16_t* wqkv = (bf16_t*)ws;                 ws += (size_t)3 * C_DIM * C_DIM * 2;
  bf16_t* wprj = (bf16_t*)ws;                 ws += (size_t)C_DIM * C_DIM * 2;
  bf16_t* wfc1 = (bf16_t*)ws;                 ws += (size_t)HID_DIM * C_DIM * 2;
  bf16_t* wfc2 = (bf16_t*)ws;                 ws += (size_t)C_DIM * HID_DIM * 2;
  float*  xres = (float*)ws;                  ws += (size_t)NPIX * C_DIM * 4;
  bf16_t* xw   = (bf16_t*)ws;                 ws += (size_t)NPIX * C_DIM * 2;  // also `tok`
  bf16_t* qbuf = (bf16_t*)ws;                 ws += (size_t)NPIX * C_DIM * 2;
  bf16_t* kbuf = (bf16_t*)ws;                 ws += (size_t)NPIX * C_DIM * 2;
  bf16_t* vbuf = (bf16_t*)ws;                 ws += (size_t)NPIX * C_DIM * 2;
  bf16_t* aout = (bf16_t*)ws;                 ws += (size_t)NPIX * C_DIM * 2;
  bf16_t* hidden = qbuf;  // q/k/v/aout (4*14.2MB) dead after proj; alias for 9216x3072

  // weight conversions
  {
    int n;
    n = 3 * C_DIM * C_DIM;  cvt_f32_bf16<<<(n + 255) / 256, 256, 0, stream>>>(qkv_w, wqkv, n);
    n = C_DIM * C_DIM;      cvt_f32_bf16<<<(n + 255) / 256, 256, 0, stream>>>(proj_w, wprj, n);
    n = HID_DIM * C_DIM;    cvt_f32_bf16<<<(n + 255) / 256, 256, 0, stream>>>(fc1_w, wfc1, n);
    n = C_DIM * HID_DIM;    cvt_f32_bf16<<<(n + 255) / 256, 256, 0, stream>>>(fc2_w, wfc2, n);
  }

  // LN1 + shift + window partition -> xw (window-row-major, bf16)
  ln_kernel<<<NPIX / 32, 256, 0, stream>>>(x_chw, n1w, n1b, xw, 1);

  // QKV GEMM (9216x768 @ 768x2304^T) -> q,k,v (v transposed per head)
  wmma_gemm<0><<<dim3(NPIX / 128, 2304 / 128), 256, 0, stream>>>(
      xw, wqkv, qkv_b, C_DIM, 3 * C_DIM, nullptr, nullptr, nullptr, qbuf, kbuf, vbuf);

  // attention: 144 windows x 24 heads
  attn_kernel<<<NWIN * NHEADS, 128, 0, stream>>>(qbuf, kbuf, vbuf, btab, aout);

  // proj GEMM + merge/unshift + residual -> xres (C,H,W f32)
  wmma_gemm<1><<<dim3(NPIX / 128, C_DIM / 128), 256, 0, stream>>>(
      aout, wprj, proj_b, C_DIM, C_DIM, xres, x_chw, nullptr, nullptr, nullptr, nullptr);

  // LN2 -> tok (plain token order, bf16)
  ln_kernel<<<NPIX / 32, 256, 0, stream>>>(xres, n2w, n2b, xw, 0);

  // fc1 + GELU -> hidden (9216x3072 bf16)
  wmma_gemm<2><<<dim3(NPIX / 128, HID_DIM / 128), 256, 0, stream>>>(
      xw, wfc1, fc1_b, C_DIM, HID_DIM, nullptr, nullptr, hidden, nullptr, nullptr, nullptr);

  // fc2 + residual -> out (C,H,W f32)
  wmma_gemm<3><<<dim3(NPIX / 128, C_DIM / 128), 256, 0, stream>>>(
      hidden, wfc2, fc2_b, HID_DIM, C_DIM, out, xres, nullptr, nullptr, nullptr, nullptr);
}
